// AIA_1_56049323213170
// MI455X (gfx1250) — matrix operand order repaired and checked by reading, not compiled, not run on hardware
//
#include <hip/hip_runtime.h>
#include <hip/hip_bf16.h>
#include <math.h>

// ---------------- CDNA5 WMMA types ----------------
typedef __attribute__((ext_vector_type(16))) __bf16 v16bf;
typedef __attribute__((ext_vector_type(8)))  float  v8f;

union FragBF {
    v16bf v;
    unsigned short u[16];
    unsigned int   d[8];
};

#define DEVI static __device__ __forceinline__

DEVI unsigned short f2bf(float f) {              // fp32 -> bf16, round-to-nearest-even
    unsigned u = __float_as_uint(f);
    unsigned r = u + 0x7FFFu + ((u >> 16) & 1u);
    return (unsigned short)(r >> 16);
}

// Problem constants
#define NB    2
#define NC    256
#define NH    128
#define NW    128
#define NX    (NB*NC*NH*NW)      // 8388608
#define KCONV 2304               // 256*9
#define NPIX2 8192               // 2*64*64
#define PLANE 4096               // 64*64

// ---------------- conversion / fold kernels ----------------
__global__ void k_f32_to_bf16(const float* __restrict__ in, unsigned short* __restrict__ out, int n) {
    int i = blockIdx.x * blockDim.x + threadIdx.x;
    if (i < n) out[i] = f2bf(in[i]);
}

__global__ void k_fold_weights(const float* __restrict__ w, const float* __restrict__ b,
                               const float* __restrict__ g, const float* __restrict__ be,
                               const float* __restrict__ m, const float* __restrict__ var,
                               unsigned short* __restrict__ wbf, float* __restrict__ bfold) {
    int i = blockIdx.x * blockDim.x + threadIdx.x;
    if (i < NC * KCONV) {
        int oc = i / KCONV;
        float scale = g[oc] * rsqrtf(var[oc] + 1e-5f);
        wbf[i] = f2bf(w[i] * scale);
    }
    if (i < NC) {
        float scale = g[i] * rsqrtf(var[i] + 1e-5f);
        bfold[i] = (b[i] - m[i]) * scale + be[i];
    }
}

// ---------------- CIA GEMM1: (256 x 32768) * (32768 x 256) -> 256 x 256 ----------------
__global__ void k_cia_gemm1(const unsigned short* __restrict__ xbf, float* __restrict__ amat) {
    const int lane = threadIdx.x & 31;
    const int wave = threadIdx.x >> 5;
    const int tile = blockIdx.x * 8 + wave;          // 256 tiles (16 x 16)
    const int mt = tile >> 4, nt = tile & 15;
    const int hl  = lane >> 4;
    const int l15 = lane & 15;
    const size_t mrow = (size_t)(mt * 16 + l15);
    const int    ncol = nt * 16 + l15;
    v8f acc = {};
    for (int k0 = 0; k0 < 32768; k0 += 32) {
        FragBF fa, fb;
        const unsigned int* pa0 = (const unsigned int*)(xbf + mrow * 32768 + k0 + hl * 16);
        const unsigned int* pa1 = pa0 + 4;
#pragma unroll
        for (int t = 0; t < 4; ++t) { fa.d[t] = pa0[t]; fa.d[4 + t] = pa1[t]; }
#pragma unroll
        for (int vv = 0; vv < 8; ++vv) {
            int k = k0 + hl * 16 + 2 * vv;
            fb.u[2 * vv]     = xbf[(size_t)k * 256 + ncol];
            fb.u[2 * vv + 1] = xbf[(size_t)(k + 1) * 256 + ncol];
        }
        acc = __builtin_amdgcn_wmma_f32_16x16x32_bf16(false, fa.v, false, fb.v, (short)0, acc, false, false);
    }
#pragma unroll
    for (int r = 0; r < 8; ++r)
        amat[(mt * 16 + r + hl * 8) * 256 + nt * 16 + l15] = acc[r];
}

// ---------------- softmax over 256-wide rows -> bf16 ----------------
__global__ void k_softmax256_to_bf(const float* __restrict__ in, unsigned short* __restrict__ out) {
    __shared__ float red[256];
    const int row = blockIdx.x, t = threadIdx.x;
    float val = in[row * 256 + t];
    red[t] = val; __syncthreads();
    for (int s = 128; s > 0; s >>= 1) { if (t < s) red[t] = fmaxf(red[t], red[t + s]); __syncthreads(); }
    float mx = red[0]; __syncthreads();
    float e = __expf(val - mx);
    red[t] = e; __syncthreads();
    for (int s = 128; s > 0; s >>= 1) { if (t < s) red[t] += red[t + s]; __syncthreads(); }
    out[row * 256 + t] = f2bf(e / red[0]);
}

// ---------------- CIA GEMM2: (256 x 256) * (256 x 32768) -> 256 x 32768 ----------------
__global__ void k_cia_gemm2(const unsigned short* __restrict__ abf, const unsigned short* __restrict__ xbf,
                            float* __restrict__ smat) {
    const int lane = threadIdx.x & 31;
    const int wave = threadIdx.x >> 5;
    const int tile = blockIdx.x * 8 + wave;          // 32768 tiles (16 mt x 2048 nt)
    const int mt = tile & 15, nt = tile >> 4;
    const int hl  = lane >> 4;
    const int l15 = lane & 15;
    const int mrow = mt * 16 + l15;
    const int ncol = nt * 16 + l15;
    v8f acc = {};
#pragma unroll
    for (int k0 = 0; k0 < 256; k0 += 32) {
        FragBF fa, fb;
        const unsigned int* pa0 = (const unsigned int*)(abf + mrow * 256 + k0 + hl * 16);
        const unsigned int* pa1 = pa0 + 4;
#pragma unroll
        for (int t = 0; t < 4; ++t) { fa.d[t] = pa0[t]; fa.d[4 + t] = pa1[t]; }
#pragma unroll
        for (int vv = 0; vv < 8; ++vv) {
            int k = k0 + hl * 16 + 2 * vv;
            fb.u[2 * vv]     = xbf[(size_t)k * 32768 + ncol];
            fb.u[2 * vv + 1] = xbf[(size_t)(k + 1) * 32768 + ncol];
        }
        acc = __builtin_amdgcn_wmma_f32_16x16x32_bf16(false, fa.v, false, fb.v, (short)0, acc, false, false);
    }
#pragma unroll
    for (int r = 0; r < 8; ++r)
        smat[(size_t)(mt * 16 + r + hl * 8) * 32768 + nt * 16 + l15] = acc[r];
}

// ---------------- softmax over 32768-wide rows (in place) ----------------
__global__ void k_softmax_32768(float* __restrict__ smat) {
    __shared__ float red[1024];
    const int row = blockIdx.x, t = threadIdx.x;
    float* p = smat + (size_t)row * 32768;
    float vals[32];
    float mx = -3.0e38f;
#pragma unroll
    for (int i = 0; i < 32; ++i) { vals[i] = p[t + i * 1024]; mx = fmaxf(mx, vals[i]); }
    red[t] = mx; __syncthreads();
    for (int s = 512; s > 0; s >>= 1) { if (t < s) red[t] = fmaxf(red[t], red[t + s]); __syncthreads(); }
    mx = red[0]; __syncthreads();
    float sum = 0.f;
#pragma unroll
    for (int i = 0; i < 32; ++i) { vals[i] = __expf(vals[i] - mx); sum += vals[i]; }
    red[t] = sum; __syncthreads();
    for (int s = 512; s > 0; s >>= 1) { if (t < s) red[t] += red[t + s]; __syncthreads(); }
    float inv = 1.0f / red[0];
#pragma unroll
    for (int i = 0; i < 32; ++i) p[t + i * 1024] = vals[i] * inv;
}

// ---------------- softmax over trailing 128 (W axis) + residual add, emit bf16 ----------------
__global__ void k_win_softmax_add(const float* __restrict__ smat, const float* __restrict__ x,
                                  unsigned short* __restrict__ xcbf) {
    __shared__ float red[128];
    const size_t base = (size_t)blockIdx.x * 128;
    const int t = threadIdx.x;
    float val = smat[base + t];
    red[t] = val; __syncthreads();
    for (int s = 64; s > 0; s >>= 1) { if (t < s) red[t] = fmaxf(red[t], red[t + s]); __syncthreads(); }
    float mx = red[0]; __syncthreads();
    float e = __expf(val - mx);
    red[t] = e; __syncthreads();
    for (int s = 64; s > 0; s >>= 1) { if (t < s) red[t] += red[t + s]; __syncthreads(); }
    xcbf[base + t] = f2bf(x[base + t] + e / red[0]);
}

// ---------------- conv3x3 s2 p1 + folded BN as implicit GEMM (bf16 WMMA) ----------------
// mode 0: out = relu            -> outF (f32) + outBf (bf16)
// mode 1: out = relu (branch)   -> outF + outBf ; sigmoid(leakyrelu(y,0.2)) -> outBf2
__global__ void k_conv_bn(const unsigned short* __restrict__ xin, const unsigned short* __restrict__ wbf,
                          const float* __restrict__ bfold, int mode,
                          float* __restrict__ outF, unsigned short* __restrict__ outBf,
                          unsigned short* __restrict__ outBf2) {
    const int lane = threadIdx.x & 31;
    const int wave = threadIdx.x >> 5;
    const int tile = blockIdx.x * 8 + wave;          // 8192 tiles (16 mt x 512 nt)
    const int mt = tile & 15, nt = tile >> 4;
    const int hl  = lane >> 4;
    const int l15 = lane & 15;
    const int mrow = mt * 16 + l15;
    const int n   = nt * 16 + l15;                   // pixel id: b*4096 + oh*64 + ow
    const int b   = n >> 12, oh = (n >> 6) & 63, ow = n & 63;
    const int ih0 = oh * 2 - 1, iw0 = ow * 2 - 1;
    v8f acc = {};
    for (int k0 = 0; k0 < KCONV; k0 += 32) {
        FragBF fa, fb;
        const unsigned int* pa0 = (const unsigned int*)(wbf + (size_t)mrow * KCONV + k0 + hl * 16);
        const unsigned int* pa1 = pa0 + 4;
#pragma unroll
        for (int t = 0; t < 4; ++t) { fa.d[t] = pa0[t]; fa.d[4 + t] = pa1[t]; }
#pragma unroll
        for (int e = 0; e < 16; ++e) {
            int k  = k0 + hl * 16 + e;
            int ic = k / 9;
            int r9 = k - ic * 9;
            int dh = r9 / 3;
            int dw = r9 - dh * 3;
            int ih = ih0 + dh, iw = iw0 + dw;
            unsigned short vload = ((unsigned)ih < 128u && (unsigned)iw < 128u)
                ? xin[(((size_t)(b * 256 + ic)) << 14) + (ih << 7) + iw] : (unsigned short)0;
            fb.u[e] = vload;
        }
        acc = __builtin_amdgcn_wmma_f32_16x16x32_bf16(false, fa.v, false, fb.v, (short)0, acc, false, false);
    }
#pragma unroll
    for (int r = 0; r < 8; ++r) {
        int oc = mt * 16 + r + hl * 8;
        float y = acc[r] + bfold[oc];
        size_t oidx = ((size_t)(b * 256 + oc)) * PLANE + oh * 64 + ow;
        float rl = fmaxf(y, 0.f);
        outF[oidx]  = rl;
        outBf[oidx] = f2bf(rl);
        if (mode == 1) {
            float lk = (y > 0.f) ? y : 0.2f * y;
            outBf2[oidx] = f2bf(1.0f / (1.0f + __expf(-lk)));
        }
    }
}

// ---------------- batched (b,c) 64x64: softmax(A · B^T) with fused epilogues ----------------
// mode 0: write f32 (att1) ; mode 1: write bf16 (att2) ; mode 2: out64 = relu(p + add1 + add2)
__global__ void k_attn(const unsigned short* __restrict__ A, const unsigned short* __restrict__ Bm,
                       int mode, float* __restrict__ outF, unsigned short* __restrict__ outBf,
                       const float* __restrict__ add1, const float* __restrict__ add2,
                       float* __restrict__ out64) {
    __shared__ float smem[64 * 64];
    const int bc = blockIdx.x;                       // 512 planes
    const size_t base = (size_t)bc * PLANE;
    const int lane = threadIdx.x & 31;
    const int wave = threadIdx.x >> 5;               // 16 waves -> 4x4 tiles
    const int ti = wave >> 2, tj = wave & 3;
    const int hl = lane >> 4, l15 = lane & 15;
    const int irow = ti * 16 + l15;                  // A row i
    const int jrow = tj * 16 + l15;                  // B operand = Bm[j,k] (A·B^T)
    v8f acc = {};
#pragma unroll
    for (int k0 = 0; k0 < 64; k0 += 32) {
        FragBF fa, fb;
        const unsigned int* pa0 = (const unsigned int*)(A + base + irow * 64 + k0 + hl * 16);
        const unsigned int* pa1 = pa0 + 4;
#pragma unroll
        for (int t = 0; t < 4; ++t) { fa.d[t] = pa0[t]; fa.d[4 + t] = pa1[t]; }
        const unsigned int* pb = (const unsigned int*)(Bm + base + jrow * 64 + k0 + hl * 16);
#pragma unroll
        for (int t = 0; t < 8; ++t) fb.d[t] = pb[t];
        acc = __builtin_amdgcn_wmma_f32_16x16x32_bf16(false, fa.v, false, fb.v, (short)0, acc, false, false);
    }
#pragma unroll
    for (int r = 0; r < 8; ++r)
        smem[(ti * 16 + r + hl * 8) * 64 + tj * 16 + l15] = acc[r];
    __syncthreads();
    if (threadIdx.x < 64) {
        const int i = threadIdx.x;
        float* rowp = smem + i * 64;
        float mx = rowp[0];
        for (int j = 1; j < 64; ++j) mx = fmaxf(mx, rowp[j]);
        float sum = 0.f;
        for (int j = 0; j < 64; ++j) { float e = __expf(rowp[j] - mx); rowp[j] = e; sum += e; }
        float inv = 1.0f / sum;
        for (int j = 0; j < 64; ++j) {
            float p = rowp[j] * inv;
            size_t idx = base + i * 64 + j;
            if (mode == 0)      outF[idx]  = p;
            else if (mode == 1) outBf[idx] = f2bf(p);
            else                out64[idx] = fmaxf(p + add1[idx] + add2[idx], 0.f);
        }
    }
}

// ---------------- bilinear 64x64 -> 128x128 (half-pixel, align_corners=False) ----------------
__global__ void k_upsample(const float* __restrict__ in, float* __restrict__ out) {
    int idx = blockIdx.x * blockDim.x + threadIdx.x;
    if (idx >= NX) return;
    int w = idx & 127;
    int h = (idx >> 7) & 127;
    int p = idx >> 14;
    float sh = h * 0.5f - 0.25f;
    float sw = w * 0.5f - 0.25f;
    int h0 = (int)floorf(sh); float fh = sh - (float)h0;
    int w0 = (int)floorf(sw); float fw = sw - (float)w0;
    int h0c = h0 < 0 ? 0 : h0;       int h1c = (h0 + 1) > 63 ? 63 : (h0 + 1);
    int w0c = w0 < 0 ? 0 : w0;       int w1c = (w0 + 1) > 63 ? 63 : (w0 + 1);
    const float* pp = in + (size_t)p * PLANE;
    float v00 = pp[h0c * 64 + w0c], v01 = pp[h0c * 64 + w1c];
    float v10 = pp[h1c * 64 + w0c], v11 = pp[h1c * 64 + w1c];
    float top = v00 + fw * (v01 - v00);
    float bot = v10 + fw * (v11 - v10);
    out[idx] = top + fh * (bot - top);
}

// ---------------- host-side orchestration ----------------
extern "C" void kernel_launch(void* const* d_in, const int* in_sizes, int n_in,
                              void* d_out, int out_size, void* d_ws, size_t ws_size,
                              hipStream_t stream) {
    const float* x     = (const float*)d_in[0];
    const float* convw = (const float*)d_in[1];
    const float* convb = (const float*)d_in[2];
    const float* gamma = (const float*)d_in[3];
    const float* beta  = (const float*)d_in[4];
    const float* mean  = (const float*)d_in[5];
    const float* var   = (const float*)d_in[6];
    float* out = (float*)d_out;

    char* ws = (char*)d_ws;
    size_t off = 0;
    auto alloc = [&](size_t bytes) -> char* {
        char* p = ws + off;
        off += (bytes + 255) & ~(size_t)255;
        return p;
    };
    unsigned short* xbf      = (unsigned short*)alloc((size_t)NX * 2);
    unsigned short* wbf      = (unsigned short*)alloc((size_t)NC * KCONV * 2);
    float*          bfold    = (float*)         alloc((size_t)NC * 4);
    float*          amat     = (float*)         alloc((size_t)NC * NC * 4);
    unsigned short* abf      = (unsigned short*)alloc((size_t)NC * NC * 2);
    float*          smat     = (float*)         alloc((size_t)NX * 4);
    unsigned short* xcbf     = (unsigned short*)alloc((size_t)NX * 2);
    float*          x12      = (float*)         alloc((size_t)NB * NC * PLANE * 4);
    unsigned short* x12bf    = (unsigned short*)alloc((size_t)NB * NC * PLANE * 2);
    float*          branch   = (float*)         alloc((size_t)NB * NC * PLANE * 4);
    unsigned short* branchbf = (unsigned short*)alloc((size_t)NB * NC * PLANE * 2);
    unsigned short* x43bf    = (unsigned short*)alloc((size_t)NB * NC * PLANE * 2);
    float*          att1     = (float*)         alloc((size_t)NB * NC * PLANE * 4);
    unsigned short* att2bf   = (unsigned short*)alloc((size_t)NB * NC * PLANE * 2);
    float*          out64    = (float*)         alloc((size_t)NB * NC * PLANE * 4);

    // 0) precision conversion + BN fold
    k_f32_to_bf16<<<NX / 256, 256, 0, stream>>>(x, xbf, NX);
    k_fold_weights<<<(NC * KCONV + 255) / 256, 256, 0, stream>>>(convw, convb, gamma, beta, mean, var, wbf, bfold);

    // 1) CIA: softmax(flat_c @ flat_r) -> softmax(a @ flat_c) -> row/window softmax + residual
    k_cia_gemm1<<<32, 256, 0, stream>>>(xbf, amat);
    k_softmax256_to_bf<<<256, 256, 0, stream>>>(amat, abf);
    k_cia_gemm2<<<4096, 256, 0, stream>>>(abf, xbf, smat);
    k_softmax_32768<<<256, 1024, 0, stream>>>(smat);
    k_win_softmax_add<<<NX / 128, 128, 0, stream>>>(smat, x, xcbf);

    // 2) conv+BN (implicit GEMM, bf16 WMMA): once on cia(x), once on x
    k_conv_bn<<<1024, 256, 0, stream>>>(xcbf, wbf, bfold, 0, x12, x12bf, (unsigned short*)nullptr);
    k_conv_bn<<<1024, 256, 0, stream>>>(xbf,  wbf, bfold, 1, branch, branchbf, x43bf);

    // 3) batched 64x64 attention matmuls + softmax, fused combine on the last
    k_attn<<<NB * NC, 512, 0, stream>>>(x12bf, branchbf, 0, att1, (unsigned short*)nullptr,
                                        (const float*)nullptr, (const float*)nullptr, (float*)nullptr);
    k_attn<<<NB * NC, 512, 0, stream>>>(branchbf, x43bf, 1, (float*)nullptr, att2bf,
                                        (const float*)nullptr, (const float*)nullptr, (float*)nullptr);
    k_attn<<<NB * NC, 512, 0, stream>>>(x12bf, att2bf, 2, (float*)nullptr, (unsigned short*)nullptr,
                                        att1, branch, out64);

    // 4) bilinear upsample to (2,256,128,128)
    k_upsample<<<NX / 256, 256, 0, stream>>>(out64, out);
}
